// HGTModel_86371792322718
// MI455X (gfx1250) — compile-verified, matrix operand order checked by reading, not run on hardware
//
#include <hip/hip_runtime.h>

#define HID 256
#define IN_DIM 768
#define HD 64
#define NH 4

#define BM 128
#define BN 128
#define KT 32
#define LPAD 8

#if __has_builtin(__builtin_amdgcn_global_load_async_to_lds_b128) && \
    __has_builtin(__builtin_amdgcn_s_wait_asynccnt)
#define USE_ASYNC_LDS 1
#else
#define USE_ASYNC_LDS 0
#endif

#if USE_ASYNC_LDS
typedef int v4i_a __attribute__((__vector_size__(4 * sizeof(int))));
#define GPTR(p) ((__attribute__((address_space(1))) v4i_a*)(p))
#define LPTR(p) ((__attribute__((address_space(3))) v4i_a*)(p))
#endif

typedef __attribute__((ext_vector_type(16))) __bf16 v16bf;
typedef __attribute__((ext_vector_type(8)))  __bf16 v8bf;
typedef __attribute__((ext_vector_type(8)))  float  v8f;

// ---------- scalar helpers ----------
__device__ inline __bf16 f2bf(float f) {
  union { float f; unsigned u; } a; a.f = f;
  unsigned r = a.u + 0x7FFFu + ((a.u >> 16) & 1u);   // round-to-nearest-even
  union { unsigned short s; __bf16 b; } o; o.s = (unsigned short)(r >> 16);
  return o.b;
}
__device__ inline float bf2f(__bf16 b) {
  union { unsigned short s; __bf16 b; } i; i.b = b;
  union { unsigned u; float f; } o; o.u = ((unsigned)i.s) << 16;
  return o.f;
}
__device__ inline float ldval(const float* p)  { return *p; }
__device__ inline float ldval(const __bf16* p) { return bf2f(*p); }

__device__ inline float gelu_exact(float x) {
  return 0.5f * x * (1.f + erff(x * 0.7071067811865475f));
}

// float <-> order-preserving uint key (for atomic segment max)
__device__ inline unsigned fkey(float f) {
  unsigned u = __float_as_uint(f);
  return (u & 0x80000000u) ? ~u : (u | 0x80000000u);
}
__device__ inline float funkey(unsigned k) {
  unsigned u = (k & 0x80000000u) ? (k ^ 0x80000000u) : ~k;
  return __uint_as_float(u);
}
#define NEGKEY 0x007FFFFFu   // fkey(-inf)

// ---------- WMMA fragment load (16-bit, 16x32 tile, documented CDNA5 layout) ----------
// lanes 0-15 : row r, K = {0..7, 16..23};  lanes 16-31 : row r, K = {8..15, 24..31}
__device__ inline v16bf load_frag16(const __bf16* base, int lane, int stride) {
  int half = lane >> 4, r = lane & 15;
  const __bf16* p = base + (long)r * stride + half * 8;
  v16bf f;
#pragma unroll
  for (int j = 0; j < 8; ++j) {
    int k = ((j & 3) << 1) + ((j >> 2) << 4);
    f[2 * j]     = p[k];
    f[2 * j + 1] = p[k + 1];
  }
  return f;
}

// ---------- generic WMMA GEMM:  C = act(A) @ Bt^T + bias, epilogues ----------
// A: [M,K] (fp32 or bf16, row stride lda), Bt: [N,K] bf16 (pre-transposed weights)
// OUT_MODE: 0 = f32, 1 = f32+ReLU, 2 = bf16, 3 = f32 sigmoid-skip mix (HGT finish)
template <typename AT, int GELU_A, int BIAS, int OUT_MODE>
__global__ __launch_bounds__(256) void gemm_wmma(
    const AT* __restrict__ A, long lda, long aBatch,
    const __bf16* __restrict__ Bt, long bBatch,
    const float* __restrict__ bias,
    float* Cf, __bf16* Cb, long ldc, long cBatch,
    const float* __restrict__ skipX, const float* __restrict__ skipLogit,
    int M, int N, int K) {
  __shared__ alignas(16) __bf16 As[BM][KT + LPAD];
  __shared__ alignas(16) __bf16 Bs[BN][KT + LPAD];

  const int tid  = threadIdx.x;
  const int wid  = tid >> 5, lane = tid & 31;
  const int wm   = wid & 3,  wn   = wid >> 2;       // 4x2 wave grid
  const int m0   = blockIdx.y * BM, n0 = blockIdx.x * BN;

  A  += (long)blockIdx.z * aBatch;
  Bt += (long)blockIdx.z * bBatch;
  const long cz = (long)blockIdx.z * cBatch;

  v8f acc[2][4];
#pragma unroll
  for (int i = 0; i < 2; ++i)
#pragma unroll
    for (int j = 0; j < 4; ++j) {
      v8f z = {0.f, 0.f, 0.f, 0.f, 0.f, 0.f, 0.f, 0.f};
      acc[i][j] = z;
    }

  for (int k0 = 0; k0 < K; k0 += KT) {
#if USE_ASYNC_LDS
    // ---- B tile via async global->LDS (ASYNCcnt path, bypasses VGPRs) ----
    // 128 rows x 64B = 512 x 16B chunks; OOB rows skipped (their outputs are
    // never stored, so garbage LDS content there is harmless).
#pragma unroll
    for (int c2 = 0; c2 < 2; ++c2) {
      int cidx = tid + c2 * 256;
      int r = cidx >> 2, q = cidx & 3;
      if (n0 + r < N)
        __builtin_amdgcn_global_load_async_to_lds_b128(
            GPTR(&Bt[(long)(n0 + r) * K + k0 + q * 8]), LPTR(&Bs[r][q * 8]), 0, 0);
    }
#endif
    if (k0 + KT < K) {   // prefetch next K tile (speculative, gfx1250 global_prefetch)
      int r = tid >> 1;
      if (m0 + r < M) __builtin_prefetch(&A[(long)(m0 + r) * lda + k0 + KT], 0, 3);
      if (n0 + r < N) __builtin_prefetch(&Bt[(long)(n0 + r) * K + k0 + KT], 0, 3);
    }

    // ---- A tile ----
    bool asyncA = false;
#if USE_ASYNC_LDS
    if constexpr (sizeof(AT) == 2 && GELU_A == 0) {
      asyncA = true;   // bf16 A needs no conversion: async copy straight to LDS
#pragma unroll
      for (int c2 = 0; c2 < 2; ++c2) {
        int cidx = tid + c2 * 256;
        int r = cidx >> 2, q = cidx & 3;
        if (m0 + r < M)
          __builtin_amdgcn_global_load_async_to_lds_b128(
              GPTR(&A[(long)(m0 + r) * lda + k0 + q * 8]), LPTR(&As[r][q * 8]), 0, 0);
      }
    }
#endif
    if (!asyncA) {      // fp32 (or gelu) A: load, activate, convert, ds_store
#pragma unroll
      for (int i = 0; i < 16; ++i) {
        int idx = tid + i * 256;
        int r = idx >> 5, c = idx & 31;
        int gm = m0 + r, gk = k0 + c;
        float v = 0.f;
        if (gm < M && gk < K) v = ldval(&A[(long)gm * lda + gk]);
        if (GELU_A) v = gelu_exact(v);
        As[r][c] = f2bf(v);
      }
    }

#if USE_ASYNC_LDS
    __builtin_amdgcn_s_wait_asynccnt(0);   // drain this wave's async copies
#else
    // fallback B staging through VGPRs
#pragma unroll
    for (int i = 0; i < 16; ++i) {
      int idx = tid + i * 256;
      int r = idx >> 5, c = idx & 31;
      int gn = n0 + r, gk = k0 + c;
      __bf16 bv = f2bf(0.f);
      if (gn < N && gk < K) bv = Bt[(long)gn * K + gk];
      Bs[r][c] = bv;
    }
#endif
    __syncthreads();

    v16bf af[2], bfr[4];
#pragma unroll
    for (int i = 0; i < 2; ++i) af[i]  = load_frag16(&As[wm * 32 + i * 16][0], lane, KT + LPAD);
#pragma unroll
    for (int j = 0; j < 4; ++j) bfr[j] = load_frag16(&Bs[wn * 64 + j * 16][0], lane, KT + LPAD);
#pragma unroll
    for (int i = 0; i < 2; ++i)
#pragma unroll
      for (int j = 0; j < 4; ++j)
        acc[i][j] = __builtin_amdgcn_wmma_f32_16x16x32_bf16(
            false, af[i], false, bfr[j], (short)0, acc[i][j], false, false);
    __syncthreads();
  }

  // store: C 16x16 f32 layout — VGPR v -> row (v | v+8), lane&15 -> col
  const int half = lane >> 4, r = lane & 15;
  float sig = 0.f;
  if (OUT_MODE == 3) sig = 1.f / (1.f + expf(-skipLogit[0]));
#pragma unroll
  for (int i = 0; i < 2; ++i)
#pragma unroll
    for (int j = 0; j < 4; ++j)
#pragma unroll
      for (int v = 0; v < 8; ++v) {
        int row = m0 + wm * 32 + i * 16 + half * 8 + v;
        int col = n0 + wn * 64 + j * 16 + r;
        if (row < M && col < N) {
          float val = acc[i][j][v];
          if (BIAS) val += bias[col];
          long o = cz + (long)row * ldc + col;
          if (OUT_MODE == 0)      Cf[o] = val;
          else if (OUT_MODE == 1) Cf[o] = fmaxf(val, 0.f);
          else if (OUT_MODE == 2) Cb[o] = f2bf(val);
          else {
            float x = skipX[(long)row * ldc + col];
            Cf[o] = sig * val + (1.f - sig) * x;
          }
        }
      }
}

// ---------- weight convert + transpose: fp32 [K,N] -> bf16 [N,K] (batched in z) ----------
__global__ void cvt_transpose_bf16(const float* __restrict__ in, __bf16* __restrict__ out,
                                   int K, int N) {
  long total = (long)K * N;
  in  += (long)blockIdx.z * total;
  out += (long)blockIdx.z * total;
  long idx = (long)blockIdx.x * blockDim.x + threadIdx.x;
  if (idx >= total) return;
  long n = idx / K, k = idx - n * K;
  out[idx] = f2bf(in[k * (long)N + n]);
}

__global__ void fill_u32_kernel(unsigned* p, unsigned v, long n) {
  long i = (long)blockIdx.x * blockDim.x + threadIdx.x;
  if (i < n) p[i] = v;
}

// ---------- edge logits: one wave per edge; 4 heads x 64-dot via 8-lane groups ----------
__global__ void edge_logits_kernel(const __bf16* __restrict__ qbase, const __bf16* __restrict__ krel,
                                   const int* __restrict__ src, const int* __restrict__ dst,
                                   const float* __restrict__ prel, float* __restrict__ logits,
                                   unsigned* __restrict__ mkey, int E, float scale) {
  int gw   = (int)(((long)blockIdx.x * blockDim.x + threadIdx.x) >> 5);
  int lane = threadIdx.x & 31;
  if (gw >= E) return;
  int s = src[gw], d = dst[gw];
  const v8bf* qp = (const v8bf*)(qbase + (long)d * (3 * HID)) + lane;
  const v8bf* kp = (const v8bf*)(krel + (long)s * HID) + lane;
  v8bf qv = *qp, kv = *kp;
  float acc = 0.f;
#pragma unroll
  for (int i = 0; i < 8; ++i) acc += bf2f(qv[i]) * bf2f(kv[i]);
  acc += __shfl_xor(acc, 1);
  acc += __shfl_xor(acc, 2);
  acc += __shfl_xor(acc, 4);
  if ((lane & 7) == 0) {
    int h = lane >> 3;
    float l = acc * scale * prel[h];
    logits[(long)gw * NH + h] = l;
    atomicMax(&mkey[(long)d * NH + h], fkey(l));
  }
}

// ---------- exp(l - max) and segment sum ----------
__global__ void exp_sum_kernel(float* __restrict__ logits, const int* __restrict__ dst,
                               const unsigned* __restrict__ mkey, float* __restrict__ ssum, int E) {
  long idx = (long)blockIdx.x * blockDim.x + threadIdx.x;
  if (idx >= (long)E * NH) return;
  int i = (int)(idx >> 2), h = (int)(idx & 3);
  int d = dst[i];
  float m = funkey(mkey[(long)d * NH + h]);
  float e = expf(logits[idx] - m);
  logits[idx] = e;
  atomicAdd(&ssum[(long)d * NH + h], e);
}

// ---------- normalized message scatter-add ----------
__global__ void scatter_kernel(const float* __restrict__ evals, const float* __restrict__ ssum,
                               const int* __restrict__ src, const int* __restrict__ dst,
                               const __bf16* __restrict__ vrel, float* __restrict__ aggr, int E) {
  long idx = (long)blockIdx.x * blockDim.x + threadIdx.x;
  if (idx >= (long)E * HID) return;
  int i = (int)(idx >> 8), c = (int)(idx & (HID - 1));
  int h = c >> 6;
  int s = src[i], d = dst[i];
  float a = evals[(long)i * NH + h] / (ssum[(long)d * NH + h] + 1e-16f);
  atomicAdd(&aggr[(long)d * HID + c], a * bf2f(vrel[(long)s * HID + c]));
}

// ---------- final classifier head: out = H @ Wc + bc ----------
__global__ void head_kernel(const float* __restrict__ Hb, const float* __restrict__ Wc,
                            const float* __restrict__ bc, float* __restrict__ out, int M) {
  int idx = blockIdx.x * blockDim.x + threadIdx.x;
  if (idx >= M * 2) return;
  int m = idx >> 1, c = idx & 1;
  float s = bc[c];
  const float* hp = Hb + (long)m * 128;
#pragma unroll 4
  for (int k = 0; k < 128; ++k) s += hp[k] * Wc[k * 2 + c];
  out[idx] = s;
}

// =====================================================================
extern "C" void kernel_launch(void* const* d_in, const int* in_sizes, int n_in,
                              void* d_out, int out_size, void* d_ws, size_t ws_size,
                              hipStream_t stream) {
  (void)n_in; (void)ws_size;
  const int NU = in_sizes[0] / IN_DIM;
  const int NM = in_sizes[1] / IN_DIM;
  const int NR = in_sizes[2] / IN_DIM;
  const int E  = in_sizes[9];
  const int BS = out_size / 2;

  // ---- workspace layout ----
  size_t off = 0;
  auto alloc = [&](size_t bytes) -> char* {
    char* p = (char*)d_ws + off;
    off = (off + bytes + 255) & ~(size_t)255;
    return p;
  };
  float*  XU0    = (float*)alloc((size_t)NU * HID * 4);
  float*  XU1    = (float*)alloc((size_t)NU * HID * 4);
  float*  XM0    = (float*)alloc((size_t)NM * HID * 4);
  float*  XR0    = (float*)alloc((size_t)NR * HID * 4);
  float*  XR1    = (float*)alloc((size_t)NR * HID * 4);
  __bf16* KQV_U  = (__bf16*)alloc((size_t)NU * 3 * HID * 2);
  __bf16* KQV_M  = (__bf16*)alloc((size_t)NM * 3 * HID * 2);
  __bf16* KQV_R  = (__bf16*)alloc((size_t)NR * 3 * HID * 2);
  __bf16* KREL0  = (__bf16*)alloc((size_t)NM * HID * 2);
  __bf16* VREL0  = (__bf16*)alloc((size_t)NM * HID * 2);
  __bf16* KREL1  = (__bf16*)alloc((size_t)NU * HID * 2);
  __bf16* VREL1  = (__bf16*)alloc((size_t)NU * HID * 2);
  __bf16* KREL2  = (__bf16*)alloc((size_t)NR * HID * 2);
  __bf16* VREL2  = (__bf16*)alloc((size_t)NR * HID * 2);
  unsigned* MKEY_R = (unsigned*)alloc((size_t)NR * NH * 4);
  float*    S_R    = (float*)alloc((size_t)NR * NH * 4);
  unsigned* MKEY_U = (unsigned*)alloc((size_t)NU * NH * 4);
  float*    S_U    = (float*)alloc((size_t)NU * NH * 4);
  float*  LOG_R  = (float*)alloc((size_t)2 * E * NH * 4);
  float*  LOG_U  = (float*)alloc((size_t)E * NH * 4);
  float*  AGGR_R = (float*)alloc((size_t)NR * HID * 4);
  float*  AGGR_U = (float*)alloc((size_t)NU * HID * 4);
  float*  HB     = (float*)alloc((size_t)BS * 128 * 4);
  __bf16* W1T    = (__bf16*)alloc((size_t)IN_DIM * HID * 2);
  __bf16* WKQVT  = (__bf16*)alloc((size_t)2 * 3 * HID * 3 * HID * 2);
  __bf16* ARELT  = (__bf16*)alloc((size_t)2 * 12 * HD * HD * 2);
  __bf16* MRELT  = (__bf16*)alloc((size_t)2 * 12 * HD * HD * 2);
  __bf16* WOUTT  = (__bf16*)alloc((size_t)2 * 3 * HID * HID * 2);
  __bf16* W2T    = (__bf16*)alloc((size_t)HID * 128 * 2);

  // ---- weight conversion (fp32 [K,N] -> bf16 [N,K]) ----
  auto cvt = [&](const float* in, __bf16* out, int K, int N, int z) {
    long total = (long)K * N;
    dim3 g((unsigned)((total + 255) / 256), 1, (unsigned)z);
    cvt_transpose_bf16<<<g, 256, 0, stream>>>(in, out, K, N);
  };
  const size_t KQVMAT = (size_t)HID * 3 * HID;  // per-type Wkqv elems
  cvt((const float*)d_in[3],  W1T, IN_DIM, HID, 1);
  cvt((const float*)d_in[16], WKQVT,                HID, 3 * HID, 3);
  cvt((const float*)d_in[24], WKQVT + 3 * KQVMAT,   HID, 3 * HID, 3);
  cvt((const float*)d_in[18], ARELT,                 HD, HD, 12);
  cvt((const float*)d_in[26], ARELT + 12 * HD * HD,  HD, HD, 12);
  cvt((const float*)d_in[19], MRELT,                 HD, HD, 12);
  cvt((const float*)d_in[27], MRELT + 12 * HD * HD,  HD, HD, 12);
  cvt((const float*)d_in[22], WOUTT,                  HID, HID, 3);
  cvt((const float*)d_in[30], WOUTT + 3 * HID * HID,  HID, HID, 3);
  cvt((const float*)d_in[5],  W2T, HID, 128, 1);

  // ---- GEMM launch helpers ----
  auto g_relu = [&](const float* A, long lda, const __bf16* Bt, const float* bias,
                    float* C, long ldc, int M, int N, int K) {
    dim3 g((unsigned)((N + BN - 1) / BN), (unsigned)((M + BM - 1) / BM), 1);
    gemm_wmma<float, 0, 1, 1><<<g, 256, 0, stream>>>(A, lda, 0, Bt, 0, bias,
        C, nullptr, ldc, 0, nullptr, nullptr, M, N, K);
  };
  auto g_bf16 = [&](const float* A, long lda, const __bf16* Bt, const float* bias,
                    __bf16* C, long ldc, int M, int N, int K) {
    dim3 g((unsigned)((N + BN - 1) / BN), (unsigned)((M + BM - 1) / BM), 1);
    gemm_wmma<float, 0, 1, 2><<<g, 256, 0, stream>>>(A, lda, 0, Bt, 0, bias,
        nullptr, C, ldc, 0, nullptr, nullptr, M, N, K);
  };
  auto g_rel = [&](const __bf16* A, const __bf16* Bt, __bf16* C, int M) {
    dim3 g(1, (unsigned)((M + BM - 1) / BM), NH);   // z = heads
    gemm_wmma<__bf16, 0, 0, 2><<<g, 256, 0, stream>>>(A, 3 * HID, HD, Bt, HD * HD,
        nullptr, nullptr, C, HID, HD, nullptr, nullptr, M, HD, HD);
  };
  auto g_finish = [&](const float* AGGR, const __bf16* WoutT, const float* bias,
                      float* C, const float* skipX, const float* slog, int M) {
    dim3 g((unsigned)((HID + BN - 1) / BN), (unsigned)((M + BM - 1) / BM), 1);
    gemm_wmma<float, 1, 1, 3><<<g, 256, 0, stream>>>(AGGR, HID, 0, WoutT, 0, bias,
        C, nullptr, HID, 0, skipX, slog, M, HID, HID);
  };

  const int* mc_src = (const int*)d_in[9];
  const int* mc_dst = (const int*)d_in[10];
  const int* up_src = (const int*)d_in[11];
  const int* up_dst = (const int*)d_in[12];
  const int* rp_src = (const int*)d_in[13];
  const int* rp_dst = (const int*)d_in[14];

  // ---- input projection: t = relu(x @ W1 + b1) ----
  const float* b1 = (const float*)d_in[4];
  g_relu((const float*)d_in[0], IN_DIM, W1T, b1, XU0, HID, NU, HID, IN_DIM);
  g_relu((const float*)d_in[1], IN_DIM, W1T, b1, XM0, HID, NM, HID, IN_DIM);
  g_relu((const float*)d_in[2], IN_DIM, W1T, b1, XR0, HID, NR, HID, IN_DIM);

  float *xin_u = XU0, *xin_r = XR0, *xout_u = XU1, *xout_r = XR1;

  for (int l = 0; l < 2; ++l) {
    const int base = 16 + l * 8;
    const float* bkqv = (const float*)d_in[base + 1];
    const float* prel = (const float*)d_in[base + 4];
    const float* skpv = (const float*)d_in[base + 5];
    const float* bout = (const float*)d_in[base + 7];
    const __bf16* WkqvTl = WKQVT + (size_t)l * 3 * KQVMAT;

    // kqv per node type (0=user, 1=movie, 2=review)
    g_bf16(xin_u, HID, WkqvTl + 0 * KQVMAT, bkqv + 0 * 3 * HID, KQV_U, 3 * HID, NU, 3 * HID, HID);
    g_bf16(XM0,   HID, WkqvTl + 1 * KQVMAT, bkqv + 1 * 3 * HID, KQV_M, 3 * HID, NM, 3 * HID, HID);
    g_bf16(xin_r, HID, WkqvTl + 2 * KQVMAT, bkqv + 2 * 3 * HID, KQV_R, 3 * HID, NR, 3 * HID, HID);

    // per-relation k/v transforms (edge src types: 0=movie, 1=user, 2=review)
    const __bf16* srcKQV[3] = {KQV_M, KQV_U, KQV_R};
    const int     nsrc[3]   = {NM, NU, NR};
    __bf16*       krel[3]   = {KREL0, KREL1, KREL2};
    __bf16*       vrel[3]   = {VREL0, VREL1, VREL2};
    for (int e = 0; e < 3; ++e) {
      const __bf16* at = ARELT + (size_t)(l * 12 + e * NH) * HD * HD;
      const __bf16* mt = MRELT + (size_t)(l * 12 + e * NH) * HD * HD;
      g_rel(srcKQV[e],            at, krel[e], nsrc[e]);      // k-section (offset 0)
      g_rel(srcKQV[e] + 2 * HID,  mt, vrel[e], nsrc[e]);      // v-section (offset 512)
    }

    // attention state init
    fill_u32_kernel<<<(unsigned)(((long)NR * NH + 255) / 256), 256, 0, stream>>>(MKEY_R, NEGKEY, (long)NR * NH);
    fill_u32_kernel<<<(unsigned)(((long)NU * NH + 255) / 256), 256, 0, stream>>>(MKEY_U, NEGKEY, (long)NU * NH);
    (void)hipMemsetAsync(S_R, 0, (size_t)NR * NH * 4, stream);
    (void)hipMemsetAsync(S_U, 0, (size_t)NU * NH * 4, stream);
    (void)hipMemsetAsync(AGGR_R, 0, (size_t)NR * HID * 4, stream);
    (void)hipMemsetAsync(AGGR_U, 0, (size_t)NU * HID * 4, stream);

    // logits + segment max   (q-section starts at offset HID)
    const float scale = 0.125f;  // 1/sqrt(64)
    unsigned lb = (unsigned)(((long)E * 32 + 255) / 256);
    edge_logits_kernel<<<lb, 256, 0, stream>>>(KQV_R + HID, KREL0, mc_src, mc_dst,
        prel + 0 * NH, LOG_R,                 MKEY_R, E, scale);
    edge_logits_kernel<<<lb, 256, 0, stream>>>(KQV_R + HID, KREL1, up_src, up_dst,
        prel + 1 * NH, LOG_R + (size_t)E * NH, MKEY_R, E, scale);
    edge_logits_kernel<<<lb, 256, 0, stream>>>(KQV_U + HID, KREL2, rp_src, rp_dst,
        prel + 2 * NH, LOG_U,                 MKEY_U, E, scale);

    // exp + segment sum (joint softmax over edge types 0,1 into review dst)
    unsigned eb = (unsigned)(((long)E * NH + 255) / 256);
    exp_sum_kernel<<<eb, 256, 0, stream>>>(LOG_R,                 mc_dst, MKEY_R, S_R, E);
    exp_sum_kernel<<<eb, 256, 0, stream>>>(LOG_R + (size_t)E * NH, up_dst, MKEY_R, S_R, E);
    exp_sum_kernel<<<eb, 256, 0, stream>>>(LOG_U,                 rp_dst, MKEY_U, S_U, E);

    // normalized message scatter
    unsigned sb = (unsigned)(((long)E * HID + 255) / 256);
    scatter_kernel<<<sb, 256, 0, stream>>>(LOG_R,                 S_R, mc_src, mc_dst, VREL0, AGGR_R, E);
    scatter_kernel<<<sb, 256, 0, stream>>>(LOG_R + (size_t)E * NH, S_R, up_src, up_dst, VREL1, AGGR_R, E);
    scatter_kernel<<<sb, 256, 0, stream>>>(LOG_U,                 S_U, rp_src, rp_dst, VREL2, AGGR_U, E);

    // finish: out = sigmoid(skip)*( gelu(aggr)@Wout + bout ) + (1-sigmoid(skip))*x
    g_finish(AGGR_U, WOUTT + (size_t)(l * 3 + 0) * HID * HID, bout + 0 * HID, xout_u, xin_u, skpv + 0, NU);
    g_finish(AGGR_R, WOUTT + (size_t)(l * 3 + 2) * HID * HID, bout + 2 * HID, xout_r, xin_r, skpv + 2, NR);

    { float* t = xin_u; xin_u = xout_u; xout_u = t; }
    { float* t = xin_r; xin_r = xout_r; xout_r = t; }
  }

  // ---- classifier: h = relu(r2[:BS] @ W2 + b2); out = h @ Wc + bc ----
  g_relu(xin_r, HID, W2T, (const float*)d_in[6], HB, 128, BS, 128, HID);
  head_kernel<<<(unsigned)((BS * 2 + 255) / 256), 256, 0, stream>>>(
      HB, (const float*)d_in[7], (const float*)d_in[8], (float*)d_out, BS);
}
